// BiMultiHeadAttention_65833258713490
// MI455X (gfx1250) — compile-verified, matrix operand order checked by reading, not run on hardware
//
#include <hip/hip_runtime.h>
#include <hip/hip_bf16.h>

// ---------------------------------------------------------------------------
// BiMultiHeadAttention for MI455X (gfx1250, wave32, WMMA bf16 16x16x32)
//
// B=8, Hv=4096, Hl=256, EMBED=2048, HEADS=8, HEAD_DIM=256, SCALING=1/16
// All GEMMs run on v_wmma_f32_16x16x32_bf16 with fp32 accumulation.
// Scratch layout (bf16 intermediates, ~410MB total in d_ws):
//   q_s, vv_s : [B*H][4096][256]   (head-split)
//   k_s, lv_s : [B*H][256][256]
//   outv_s    : [B][4096][2048]
//   outl_s    : [B][256][2048]
//   colmax    : [B*H][256] f32
// ---------------------------------------------------------------------------

typedef __attribute__((ext_vector_type(16))) __bf16 v16bf;
typedef __attribute__((ext_vector_type(8)))  __bf16 v8bf;
typedef __attribute__((ext_vector_type(4)))  __bf16 v4bf;
typedef __attribute__((ext_vector_type(8)))  float  v8f;

__device__ __forceinline__ v8f wmma_bf16(v16bf a, v16bf b, v8f c) {
  return __builtin_amdgcn_wmma_f32_16x16x32_bf16(false, a, false, b,
                                                 (short)0, c, false, false);
}

// Load a 16x32 bf16 A-fragment (or transposed-B fragment) from a row-major
// tile with row stride ldr (elements). Per ISA 7.12.2:
//   lane l : row m = l&15 ; k = {8*(l>>4) + 0..7 , 8*(l>>4) + 16..23}
__device__ __forceinline__ v16bf load_frag_b(const __bf16* tile, int ldr) {
  const int lane = threadIdx.x & 31;
  const __bf16* p = tile + (size_t)(lane & 15) * ldr + ((lane >> 4) << 3);
  v8bf lo = *(const v8bf*)p;
  v8bf hi = *(const v8bf*)(p + 16);
  return __builtin_shufflevector(lo, hi, 0, 1, 2, 3, 4, 5, 6, 7,
                                 8, 9, 10, 11, 12, 13, 14, 15);
}

__device__ __forceinline__ void atomicMaxF(float* addr, float val) {
  unsigned int* ua = (unsigned int*)addr;
  unsigned int cur = __float_as_uint(*addr);
  while (__uint_as_float(cur) < val) {
    unsigned int prev = atomicCAS(ua, cur, __float_as_uint(val));
    if (prev == cur) break;
    cur = prev;
  }
}

__global__ void fill_f32(float* p, int n, float v) {
  int i = blockIdx.x * blockDim.x + threadIdx.x;
  if (i < n) p[i] = v;
}

// ---------------------------------------------------------------------------
// Projection: Y = (X[MxK] @ W[Kx2048] + bias) * scale, stored head-split bf16
//   Y[((b*8+h)*S + s)*256 + d],  b = row>>sshift, s = row & (S-1)
// Block: 256 thr = 8 waves, tile 128x64, waves 4x2 each 32x32 (2x2 frags)
// ---------------------------------------------------------------------------
__global__ __launch_bounds__(256) void proj_headsplit(
    const float* __restrict__ X, const float* __restrict__ W,
    const float* __restrict__ bias, __bf16* __restrict__ Y,
    int M, int K, int sshift, float scale) {
  constexpr int LDA = 32 + 8;
  __shared__ __align__(16) __bf16 sA[128 * LDA];
  __shared__ __align__(16) __bf16 sB[64 * LDA];  // W^T tile: [n][k]
  const int bx = blockIdx.x;
  const int m0 = (bx >> 5) * 128;   // 2048/64 = 32 n-tiles
  const int n0 = (bx & 31) * 64;
  const int tid = threadIdx.x, wid = tid >> 5, lane = tid & 31;
  const int wm = (wid & 3) * 32, wn = (wid >> 2) * 32;
  v8f acc[2][2] = {};
  for (int k0 = 0; k0 < K; k0 += 32) {
    if (k0 + 32 < K) {  // prefetch next K tile (global_prefetch_b8)
      __builtin_prefetch(&X[(size_t)(m0 + (tid >> 1)) * K + k0 + 32 +
                            ((tid & 1) << 4)], 0, 1);
      __builtin_prefetch(&W[(size_t)(k0 + 32 + (tid >> 3)) * 2048 + n0 +
                            ((tid & 7) << 3)], 0, 1);
    }
    __syncthreads();
#pragma unroll
    for (int i = 0; i < 4; ++i) {               // 128x32 f32 -> bf16 (float4)
      int idx = tid + 256 * i;                  // 1024 quads
      int r = idx >> 3, c = (idx & 7) << 2;
      float4 x = *(const float4*)&X[(size_t)(m0 + r) * K + k0 + c];
      v4bf p;
      p[0] = (__bf16)x.x; p[1] = (__bf16)x.y;
      p[2] = (__bf16)x.z; p[3] = (__bf16)x.w;
      *(v4bf*)&sA[r * LDA + c] = p;
    }
#pragma unroll
    for (int i = 0; i < 2; ++i) {               // 32x64 f32 -> bf16 ^T (float4)
      int idx = tid + 256 * i;                  // 512 quads
      int kk = idx >> 4, nq = (idx & 15) << 2;
      float4 x = *(const float4*)&W[(size_t)(k0 + kk) * 2048 + n0 + nq];
      sB[(nq + 0) * LDA + kk] = (__bf16)x.x;
      sB[(nq + 1) * LDA + kk] = (__bf16)x.y;
      sB[(nq + 2) * LDA + kk] = (__bf16)x.z;
      sB[(nq + 3) * LDA + kk] = (__bf16)x.w;
    }
    __syncthreads();
    v16bf a0 = load_frag_b(&sA[(wm) * LDA], LDA);
    v16bf a1 = load_frag_b(&sA[(wm + 16) * LDA], LDA);
    v16bf b0 = load_frag_b(&sB[(wn) * LDA], LDA);
    v16bf b1 = load_frag_b(&sB[(wn + 16) * LDA], LDA);
    acc[0][0] = wmma_bf16(a0, b0, acc[0][0]);
    acc[0][1] = wmma_bf16(a0, b1, acc[0][1]);
    acc[1][0] = wmma_bf16(a1, b0, acc[1][0]);
    acc[1][1] = wmma_bf16(a1, b1, acc[1][1]);
  }
  const int S = 1 << sshift, smask = S - 1;
#pragma unroll
  for (int mi = 0; mi < 2; ++mi)
#pragma unroll
    for (int ni = 0; ni < 2; ++ni) {
      int n = n0 + wn + ni * 16 + (lane & 15);
      int hh = n >> 8, dd = n & 255;
      float bi = bias[n];
      int mbase = m0 + wm + mi * 16 + ((lane >> 4) << 3);
#pragma unroll
      for (int i = 0; i < 8; ++i) {
        int m = mbase + i;
        int bb = m >> sshift, ss = m & smask;
        float val = (acc[mi][ni][i] + bi) * scale;
        Y[(((size_t)(bb * 8 + hh)) * S + ss) * 256 + dd] = (__bf16)val;
      }
    }
}

// ---------------------------------------------------------------------------
// Pass 1: per (b,h, 128 q rows). Each wave owns 16 q rows.
//   scores s[16x256] = q @ k^T  (WMMA) -> masks -> colmax atomics
//   -> row softmax (shuffle reductions) -> P via LDS -> out_v = P @ l_value
// ---------------------------------------------------------------------------
__global__ __launch_bounds__(256) void attn_pass1(
    const __bf16* __restrict__ q_s, const __bf16* __restrict__ k_s,
    const __bf16* __restrict__ lv_s, const int* __restrict__ mask_v,
    const int* __restrict__ mask_l, float* __restrict__ colmax,
    __bf16* __restrict__ outv_s) {
  constexpr int LDK = 256 + 8;
  __shared__ __align__(16) __bf16 sk[64 * LDK];           // k chunk / lvT chunk
  __shared__ __align__(16) __bf16 sp[8 * 16 * LDK];       // per-wave P tiles
  const int bx = blockIdx.x;
  const int bh = bx >> 5;                                 // 32 q-blocks per bh
  const int qblk = bx & 31;
  const int b = bh >> 3, h = bh & 7;
  const int tid = threadIdx.x, wid = tid >> 5, lane = tid & 31;
  const int q_base = qblk * 128 + wid * 16;
  const __bf16* qrow = q_s + ((size_t)bh * 4096 + q_base) * 256;

  // hoist all 8 q A-fragments (reused across all 4 l chunks)
  v16bf aq[8];
#pragma unroll
  for (int ks = 0; ks < 8; ++ks) aq[ks] = load_frag_b(qrow + ks * 32, 256);

  v8f sc[16] = {};                                        // n = 0..255
#pragma unroll
  for (int lc = 0; lc < 4; ++lc) {                        // l chunks of 64
    const __bf16* ksrc = k_s + ((size_t)bh * 256 + lc * 64) * 256;
    if (lc < 3)
      __builtin_prefetch(&ksrc[(size_t)(64 + (tid >> 2)) * 256 +
                               ((tid & 3) << 6)], 0, 1);
    __syncthreads();
#pragma unroll
    for (int i = 0; i < 8; ++i) {                         // 64x256 bf16 copy
      int chunk = tid + 256 * i;
      int r = chunk >> 5;
      int c = (chunk & 31) << 3;
      *(v8bf*)&sk[r * LDK + c] = *(const v8bf*)&ksrc[r * 256 + c];
    }
    __syncthreads();
#pragma unroll
    for (int ks = 0; ks < 8; ++ks) {                      // K = HEAD_DIM = 256
#pragma unroll
      for (int ni = 0; ni < 4; ++ni) {
        v16bf bb = load_frag_b(&sk[(ni * 16) * LDK + ks * 32], LDK);
        sc[lc * 4 + ni] = wmma_bf16(aq[ks], bb, sc[lc * 4 + ni]);
      }
    }
  }
  // masks (NEG = -10000 like reference)
  bool mvok[8];
#pragma unroll
  for (int i = 0; i < 8; ++i)
    mvok[i] = mask_v[b * 4096 + q_base + i + ((lane >> 4) << 3)] != 0;
#pragma unroll
  for (int t = 0; t < 16; ++t) {
    int n = t * 16 + (lane & 15);
    bool ml = mask_l[b * 256 + n] != 0;
#pragma unroll
    for (int i = 0; i < 8; ++i)
      if (!(ml && mvok[i])) sc[t][i] = -10000.0f;
  }
  // per-column (over Hv) max for pass 2
#pragma unroll
  for (int t = 0; t < 16; ++t) {
    float mx = sc[t][0];
#pragma unroll
    for (int i = 1; i < 8; ++i) mx = fmaxf(mx, sc[t][i]);
    atomicMaxF(&colmax[bh * 256 + t * 16 + (lane & 15)], mx);
  }
  // row softmax: rows live in 16-lane halves (lanes 0-15: m=i, 16-31: m=i+8)
  float rinv[8];
#pragma unroll
  for (int i = 0; i < 8; ++i) {
    float m = sc[0][i];
#pragma unroll
    for (int t = 1; t < 16; ++t) m = fmaxf(m, sc[t][i]);
#pragma unroll
    for (int off = 8; off >= 1; off >>= 1) m = fmaxf(m, __shfl_xor(m, off, 32));
    float s = 0.0f;
#pragma unroll
    for (int t = 0; t < 16; ++t) {
      float e = __expf(sc[t][i] - m);
      sc[t][i] = e;
      s += e;
    }
#pragma unroll
    for (int off = 8; off >= 1; off >>= 1) s += __shfl_xor(s, off, 32);
    rinv[i] = 1.0f / s;
  }
  // P -> LDS in A-fragment (row-major) layout
  __bf16* pw = &sp[wid * 16 * LDK];
#pragma unroll
  for (int t = 0; t < 16; ++t)
#pragma unroll
    for (int i = 0; i < 8; ++i)
      pw[(i + ((lane >> 4) << 3)) * LDK + t * 16 + (lane & 15)] =
          (__bf16)(sc[t][i] * rinv[i]);

  // hoist all 8 P A-fragments (reused across all 8 d chunks; same-wave LDS
  // ordering guarantees the stores above are visible)
  v16bf ap[8];
#pragma unroll
  for (int ks = 0; ks < 8; ++ks) ap[ks] = load_frag_b(pw + ks * 32, LDK);

  // out_v = P[16x256] @ l_value[256x256]
  v8f oacc[16] = {};
  const __bf16* lvbase = lv_s + (size_t)bh * 256 * 256;
#pragma unroll
  for (int dc = 0; dc < 8; ++dc) {                        // d chunks of 32
    __syncthreads();
#pragma unroll
    for (int i = 0; i < 4; ++i) {                         // lvT[d][kl] staging
      int idx = tid + 256 * i;                            // 1024 v8 chunks
      int kl = idx >> 2;                                  // 4 chunks per row
      int d8 = (idx & 3) << 3;
      v8bf x = *(const v8bf*)&lvbase[(size_t)kl * 256 + dc * 32 + d8];
#pragma unroll
      for (int j = 0; j < 8; ++j) sk[(d8 + j) * LDK + kl] = x[j];
    }
    __syncthreads();
#pragma unroll
    for (int ks = 0; ks < 8; ++ks) {                      // K = Hl = 256
#pragma unroll
      for (int ni = 0; ni < 2; ++ni) {
        v16bf bb = load_frag_b(&sk[(ni * 16) * LDK + ks * 32], LDK);
        oacc[dc * 2 + ni] = wmma_bf16(ap[ks], bb, oacc[dc * 2 + ni]);
      }
    }
  }
#pragma unroll
  for (int t = 0; t < 16; ++t) {
    int d = t * 16 + (lane & 15);
    size_t col = (size_t)h * 256 + d;
#pragma unroll
    for (int i = 0; i < 8; ++i) {
      int qr = q_base + i + ((lane >> 4) << 3);
      outv_s[((size_t)b * 4096 + qr) * 2048 + col] = (__bf16)oacc[t][i];
    }
  }
}

// ---------------------------------------------------------------------------
// Pass 2: per (b,h, 64 l rows). Loop all 4096 q in chunks of 32:
//   recompute scores (WMMA) -> w = exp(s - colmax) -> colsum (LDS atomics)
//   -> out_l += w^T @ v_value (WMMA) ; normalize at end.
// ---------------------------------------------------------------------------
__global__ __launch_bounds__(256) void attn_pass2(
    const __bf16* __restrict__ q_s, const __bf16* __restrict__ k_s,
    const __bf16* __restrict__ vv_s, const int* __restrict__ mask_v,
    const int* __restrict__ mask_l, const float* __restrict__ colmax,
    __bf16* __restrict__ outl_s) {
  constexpr int LDK = 256 + 8;
  constexpr int LDQ = 32 + 8;
  __shared__ __align__(16) __bf16 skl[64 * LDK];          // k rows for l range
  __shared__ __align__(16) __bf16 sq[32 * LDK];           // q chunk
  __shared__ __align__(16) __bf16 svT[256 * LDQ];         // v_value^T chunk
  __shared__ __align__(16) __bf16 swT[64 * LDQ];          // w^T chunk
  __shared__ float scolsum[64];
  __shared__ int smv[32];
  const int bx = blockIdx.x;
  const int bh = bx >> 2;
  const int lblk = bx & 3;
  const int b = bh >> 3, h = bh & 7;
  const int m0 = lblk * 64;
  const int tid = threadIdx.x, wid = tid >> 5, lane = tid & 31;

  const __bf16* ksrc = k_s + ((size_t)bh * 256 + m0) * 256;
#pragma unroll
  for (int i = 0; i < 8; ++i) {                           // stage k once
    int chunk = tid + 256 * i;
    int r = chunk >> 5;
    int c = (chunk & 31) << 3;
    *(v8bf*)&skl[r * LDK + c] = *(const v8bf*)&ksrc[r * 256 + c];
  }
  if (tid < 64) scolsum[tid] = 0.0f;
  __syncthreads();

  const int qt = wid & 1, lt = wid >> 1;                  // score tile of wave
  const int nl = lt * 16 + (lane & 15);                   // local l column
  const float cm = colmax[bh * 256 + m0 + nl];
  const bool mlok = mask_l[b * 256 + m0 + nl] != 0;

  // hoist the 8 k B-fragments: invariant across the whole q loop
  v16bf kf[8];
#pragma unroll
  for (int ks = 0; ks < 8; ++ks)
    kf[ks] = load_frag_b(&skl[(lt * 16) * LDK + ks * 32], LDK);

  v8f oacc[4][2] = {};
  const __bf16* qb = q_s + (size_t)bh * 4096 * 256;
  const __bf16* vb = vv_s + (size_t)bh * 4096 * 256;

  for (int q0 = 0; q0 < 4096; q0 += 32) {
    if (q0 + 32 < 4096) {  // prefetch next q / v chunk
      __builtin_prefetch(&qb[(size_t)(q0 + 32 + (tid >> 3)) * 256 +
                             ((tid & 7) << 5)], 0, 1);
      __builtin_prefetch(&vb[(size_t)(q0 + 32 + (tid >> 3)) * 256 +
                             ((tid & 7) << 5)], 0, 1);
    }
    __syncthreads();
#pragma unroll
    for (int i = 0; i < 4; ++i) {                         // stage q chunk
      int chunk = tid + 256 * i;
      int r = chunk >> 5;
      int c = (chunk & 31) << 3;
      *(v8bf*)&sq[r * LDK + c] = *(const v8bf*)&qb[(size_t)(q0 + r) * 256 + c];
    }
#pragma unroll
    for (int i = 0; i < 4; ++i) {                         // stage v^T chunk
      int idx = tid + 256 * i;                            // 1024 v8 chunks
      int r = idx >> 5;                                   // q row
      int d8 = (idx & 31) << 3;
      v8bf x = *(const v8bf*)&vb[(size_t)(q0 + r) * 256 + d8];
#pragma unroll
      for (int j = 0; j < 8; ++j) svT[(d8 + j) * LDQ + r] = x[j];
    }
    if (tid < 32) smv[tid] = mask_v[b * 4096 + q0 + tid];
    __syncthreads();

    v8f s = {};
#pragma unroll
    for (int ks = 0; ks < 8; ++ks) {                      // s = q @ k^T
      v16bf a = load_frag_b(&sq[(qt * 16) * LDK + ks * 32], LDK);
      s = wmma_bf16(a, kf[ks], s);
    }
    float part = 0.0f;
#pragma unroll
    for (int i = 0; i < 8; ++i) {
      int qloc = qt * 16 + i + ((lane >> 4) << 3);
      float v = s[i];
      if (!mlok || smv[qloc] == 0) v = -10000.0f;
      float w = __expf(v - cm);
      part += w;
      swT[nl * LDQ + qloc] = (__bf16)w;                   // transposed
    }
    atomicAdd(&scolsum[nl], part);
    __syncthreads();

    const int d0 = wid * 32;                              // wave's d range
    v16bf bb0 = load_frag_b(&svT[(d0) * LDQ], LDQ);
    v16bf bb1 = load_frag_b(&svT[(d0 + 16) * LDQ], LDQ);
#pragma unroll
    for (int mt = 0; mt < 4; ++mt) {
      v16bf a = load_frag_b(&swT[(mt * 16) * LDQ], LDQ);
      oacc[mt][0] = wmma_bf16(a, bb0, oacc[mt][0]);
      oacc[mt][1] = wmma_bf16(a, bb1, oacc[mt][1]);
    }
  }
  __syncthreads();
  const int d0 = wid * 32;
#pragma unroll
  for (int mt = 0; mt < 4; ++mt)
#pragma unroll
    for (int nt = 0; nt < 2; ++nt) {
      int n = d0 + nt * 16 + (lane & 15);
#pragma unroll
      for (int i = 0; i < 8; ++i) {
        int mloc = mt * 16 + i + ((lane >> 4) << 3);
        float inv = 1.0f / scolsum[mloc];
        outl_s[((size_t)b * 256 + m0 + mloc) * 2048 + h * 256 + n] =
            (__bf16)(oacc[mt][nt][i] * inv);
      }
    }
}

// ---------------------------------------------------------------------------
// Output projection: Y[MxN] = X_bf16[Mx2048] @ W[2048xN] + bias (fp32 out)
// ---------------------------------------------------------------------------
__global__ __launch_bounds__(256) void out_proj(
    const __bf16* __restrict__ X, const float* __restrict__ W,
    const float* __restrict__ bias, float* __restrict__ Y, int M, int N) {
  constexpr int LDA = 32 + 8;
  __shared__ __align__(16) __bf16 sA[128 * LDA];
  __shared__ __align__(16) __bf16 sB[64 * LDA];
  const int ntiles = N >> 6;
  const int bx = blockIdx.x;
  const int m0 = (bx / ntiles) * 128;
  const int n0 = (bx % ntiles) * 64;
  const int tid = threadIdx.x, wid = tid >> 5, lane = tid & 31;
  const int wm = (wid & 3) * 32, wn = (wid >> 2) * 32;
  v8f acc[2][2] = {};
  for (int k0 = 0; k0 < 2048; k0 += 32) {
    if (k0 + 32 < 2048) {
      __builtin_prefetch(&X[(size_t)(m0 + (tid >> 1)) * 2048 + k0 + 32 +
                            ((tid & 1) << 4)], 0, 1);
      __builtin_prefetch(&W[(size_t)(k0 + 32 + (tid >> 3)) * N + n0 +
                            ((tid & 7) << 3)], 0, 1);
    }
    __syncthreads();
#pragma unroll
    for (int i = 0; i < 2; ++i) {                         // 128x32 bf16 copy
      int chunk = tid + 256 * i;
      int r = chunk >> 2;
      int c = (chunk & 3) << 3;
      *(v8bf*)&sA[r * LDA + c] = *(const v8bf*)&X[(size_t)(m0 + r) * 2048 + k0 + c];
    }
#pragma unroll
    for (int i = 0; i < 2; ++i) {                         // 32xN f32 -> bf16 ^T
      int idx = tid + 256 * i;                            // 512 quads
      int kk = idx >> 4, nq = (idx & 15) << 2;
      float4 x = *(const float4*)&W[(size_t)(k0 + kk) * N + n0 + nq];
      sB[(nq + 0) * LDA + kk] = (__bf16)x.x;
      sB[(nq + 1) * LDA + kk] = (__bf16)x.y;
      sB[(nq + 2) * LDA + kk] = (__bf16)x.z;
      sB[(nq + 3) * LDA + kk] = (__bf16)x.w;
    }
    __syncthreads();
    v16bf a0 = load_frag_b(&sA[(wm) * LDA], LDA);
    v16bf a1 = load_frag_b(&sA[(wm + 16) * LDA], LDA);
    v16bf b0 = load_frag_b(&sB[(wn) * LDA], LDA);
    v16bf b1 = load_frag_b(&sB[(wn + 16) * LDA], LDA);
    acc[0][0] = wmma_bf16(a0, b0, acc[0][0]);
    acc[0][1] = wmma_bf16(a0, b1, acc[0][1]);
    acc[1][0] = wmma_bf16(a1, b0, acc[1][0]);
    acc[1][1] = wmma_bf16(a1, b1, acc[1][1]);
  }
#pragma unroll
  for (int mi = 0; mi < 2; ++mi)
#pragma unroll
    for (int ni = 0; ni < 2; ++ni) {
      int n = n0 + wn + ni * 16 + (lane & 15);
      float bi = bias[n];
      int mbase = m0 + wm + mi * 16 + ((lane >> 4) << 3);
#pragma unroll
      for (int i = 0; i < 8; ++i)
        Y[(size_t)(mbase + i) * N + n] = acc[mi][ni][i] + bi;
    }
}

// ---------------------------------------------------------------------------
extern "C" void kernel_launch(void* const* d_in, const int* in_sizes, int n_in,
                              void* d_out, int out_size, void* d_ws,
                              size_t ws_size, hipStream_t stream) {
  (void)in_sizes; (void)n_in; (void)out_size; (void)ws_size;
  const float* v      = (const float*)d_in[0];
  const float* l      = (const float*)d_in[1];
  const int* mask_v   = (const int*)d_in[2];
  const int* mask_l   = (const int*)d_in[3];
  const float* Wq_v   = (const float*)d_in[4];
  const float* bq_v   = (const float*)d_in[5];
  const float* Wk_l   = (const float*)d_in[6];
  const float* bk_l   = (const float*)d_in[7];
  const float* Wval_v = (const float*)d_in[8];
  const float* bval_v = (const float*)d_in[9];
  const float* Wval_l = (const float*)d_in[10];
  const float* bval_l = (const float*)d_in[11];
  const float* Wo_v   = (const float*)d_in[12];
  const float* bo_v   = (const float*)d_in[13];
  const float* Wo_l   = (const float*)d_in[14];
  const float* bo_l   = (const float*)d_in[15];
  float* out = (float*)d_out;

  char* ws = (char*)d_ws;
  size_t off = 0;
  auto carve = [&](size_t bytes) -> void* {
    void* p = ws + off;
    off += (bytes + 255) & ~(size_t)255;
    return p;
  };
  __bf16* q_s    = (__bf16*)carve((size_t)64 * 4096 * 256 * 2);   // 128MB
  __bf16* vv_s   = (__bf16*)carve((size_t)64 * 4096 * 256 * 2);   // 128MB
  __bf16* k_s    = (__bf16*)carve((size_t)64 * 256 * 256 * 2);    // 8MB
  __bf16* lv_s   = (__bf16*)carve((size_t)64 * 256 * 256 * 2);    // 8MB
  __bf16* outv_s = (__bf16*)carve((size_t)8 * 4096 * 2048 * 2);   // 128MB
  __bf16* outl_s = (__bf16*)carve((size_t)8 * 256 * 2048 * 2);    // 8MB
  float*  colmax = (float*)carve((size_t)64 * 256 * 4);

  const float SCALING = 0.0625f;  // 256^-0.5

  fill_f32<<<64, 256, 0, stream>>>(colmax, 64 * 256, -3.0e38f);

  // projections: (M/128) * (2048/64) blocks
  proj_headsplit<<<8192, 256, 0, stream>>>(v, Wq_v, bq_v, q_s, 32768, 256, 12, SCALING);
  proj_headsplit<<<512,  256, 0, stream>>>(l, Wk_l, bk_l, k_s, 2048, 768, 8, 1.0f);
  proj_headsplit<<<8192, 256, 0, stream>>>(v, Wval_v, bval_v, vv_s, 32768, 256, 12, 1.0f);
  proj_headsplit<<<512,  256, 0, stream>>>(l, Wval_l, bval_l, lv_s, 2048, 768, 8, 1.0f);

  attn_pass1<<<64 * 32, 256, 0, stream>>>(q_s, k_s, lv_s, mask_v, mask_l,
                                          colmax, outv_s);
  attn_pass2<<<64 * 4, 256, 0, stream>>>(q_s, k_s, vv_s, mask_v, mask_l,
                                         colmax, outl_s);

  out_proj<<<1024, 256, 0, stream>>>(outv_s, Wo_v, bo_v, out, 32768, 256);
  out_proj<<<192, 256, 0, stream>>>(outl_s, Wo_l, bo_l, out + (size_t)8 * 4096 * 256,
                                    2048, 768);
}
